// CascadedAttention_82394652606610
// MI455X (gfx1250) — matrix operand order repaired and checked by reading, not compiled, not run on hardware
//
#include <hip/hip_runtime.h>

#define BB 8
#define TT 128
#define DD 512
#define OO 512
#define NT16 (OO / 16)   // 32 n-tiles for all weight matrices

typedef __attribute__((ext_vector_type(16))) __bf16 v16bf;
typedef __attribute__((ext_vector_type(8)))  __bf16 v8bf;
typedef __attribute__((ext_vector_type(8)))  float  v8f;

// ---------------- wave32 reductions ----------------
__device__ inline float wave_sum(float v) {
#pragma unroll
  for (int o = 16; o > 0; o >>= 1) v += __shfl_xor(v, o, 32);
  return v;
}
__device__ inline float wave_max(float v) {
#pragma unroll
  for (int o = 16; o > 0; o >>= 1) v = fmaxf(v, __shfl_xor(v, o, 32));
  return v;
}

// ---------------- CDNA5 async global->LDS copy (ASYNCcnt path) ----------------
// 16B per lane; lds_off is the byte offset in the LDS aperture (low 32 bits of
// the generic pointer, per ISA 10.2 flat->LDS mapping). GV mode (saddr = off).
__device__ inline void async_copy_b128(unsigned lds_off, const void* gptr) {
  asm volatile("global_load_async_to_lds_b128 %0, %1, off"
               :: "v"(lds_off), "v"(gptr) : "memory");
}
__device__ inline void wait_async0() {
  asm volatile("s_wait_asynccnt 0" ::: "memory");
}
__device__ inline unsigned lds_off_of(const void* shared_ptr) {
  return (unsigned)(unsigned long long)shared_ptr;
}

// ---------------- WMMA fragment loaders ----------------
// A fragment (16x32 bf16): lane L holds row (L&15); K-chunks [(L>>4)*8, +8) and +16.
__device__ inline v16bf load_a_frag(const __bf16* rowp, int k0, int lane) {
  const int kb = (lane >> 4) * 8;
  const __bf16* p = rowp + k0 + kb;          // 16B aligned
  v8bf lo = *(const v8bf*)(p);
  v8bf hi = *(const v8bf*)(p + 16);
  return __builtin_shufflevector(lo, hi, 0,1,2,3,4,5,6,7,8,9,10,11,12,13,14,15);
}
// B fragment from pre-packed layout: chunk index ((kt*NT16 + nt)*32 + lane), 16
// contiguous halves per lane -> two contiguous 16B loads, wave reads 1KB contiguous.
__device__ inline v16bf load_b_packed(const __bf16* Bp, int k0, int nt, int lane) {
  const __bf16* p = Bp + ((((size_t)(k0 >> 5) * NT16 + nt) << 5) + lane) * 16;
  v8bf lo = *(const v8bf*)(p);
  v8bf hi = *(const v8bf*)(p + 8);
  return __builtin_shufflevector(lo, hi, 0,1,2,3,4,5,6,7,8,9,10,11,12,13,14,15);
}
__device__ inline v8f wmma_bf16(v16bf a, v16bf b, v8f c) {
  return __builtin_amdgcn_wmma_f32_16x16x32_bf16(false, a, false, b, (short)0, c,
                                                 false, false);
}

// ---------------- utility kernels ----------------
__global__ __launch_bounds__(256) void k_cvt(const float* __restrict__ src,
                                             __bf16* __restrict__ dst, int n) {
  for (int i = blockIdx.x * 256 + threadIdx.x; i < n; i += gridDim.x * 256)
    dst[i] = (__bf16)src[i];
}

__global__ __launch_bounds__(256) void k_zero(float* __restrict__ p, int n) {
  for (int i = threadIdx.x; i < n; i += 256) p[i] = 0.0f;
}

// Pack a KxN f32 weight matrix (row-major, N=512) into WMMA-B fragment order:
// lane L of tile (kt,nt): halves j = B[kt*32 + (L>>4)*8 + j + (j>=8 ? 8 : 0)][nt*16 + (L&15)], j<16.
__global__ __launch_bounds__(256) void k_pack(const float* __restrict__ src,
                                              __bf16* __restrict__ dst, int K) {
  const int idx = blockIdx.x * 256 + threadIdx.x;   // (K/32)*NT16*32 chunks
  const int total = (K >> 5) * NT16 * 32;
  if (idx >= total) return;
  const int lane = idx & 31;
  const int tile = idx >> 5;
  const int nt = tile % NT16;
  const int kt = tile / NT16;
  const int n = (nt << 4) + (lane & 15);
  const int kbase = (kt << 5) + ((lane >> 4) << 3);
  __bf16* o = dst + (size_t)idx * 16;
#pragma unroll
  for (int j = 0; j < 16; ++j) o[j] = (__bf16)src[(size_t)(kbase + j + ((j >> 3) << 3)) * OO + n];
}

// ---------------- precompute: UaH = inputs@Ua + Ba ; out = roll(inputs)@Uo + Bo ----
__global__ __launch_bounds__(256) void k_pre(const __bf16* __restrict__ inp_bf,
    const __bf16* __restrict__ UaP, const __bf16* __restrict__ UoP,
    const float* __restrict__ Ba, const float* __restrict__ Bo,
    float* __restrict__ UaH, float* __restrict__ outB) {
  const int lane = threadIdx.x & 31, w = threadIdx.x >> 5;
  const int g = blockIdx.x * 8 + w;        // 0..4095
  const int gemm = g >> 11;
  const int tix = g & 2047;
  const int mt = tix >> 5, nt = tix & 31;
  const int m0 = mt << 4;
  const int r_abs = m0 + (lane & 15);
  int src = r_abs;
  if (gemm) {                               // rolled rows: t -> (t-1) mod T
    const int b = r_abs >> 7, t = r_abs & 127;
    src = (b << 7) | ((t + 127) & 127);
  }
  const __bf16* Arow = inp_bf + (size_t)src * DD;
  const __bf16* Bp = gemm ? UoP : UaP;
  const int n = (nt << 4) + (lane & 15);
  v8f acc = {};
#pragma unroll 4
  for (int k0 = 0; k0 < DD; k0 += 32) {
    v16bf a = load_a_frag(Arow, k0, lane);
    v16bf b = load_b_packed(Bp, k0, nt, lane);
    acc = wmma_bf16(a, b, acc);
  }
  const float bv = (gemm ? Bo : Ba)[n];
  float* Dst = gemm ? outB : UaH;
#pragma unroll
  for (int i = 0; i < 8; ++i) {
    const int m = m0 + ((lane >> 4) << 3) + i;
    Dst[(size_t)m * OO + n] = acc[i] + bv;
  }
}

// ---------------- S1: s=sigmoid(pred), p=softmax(pred); WaS=s@Wa; embOut=p@emb ----
// 8 blocks x 8 waves. Blocks 0-3 -> sigmoid/Wa, 4-7 -> softmax/emb.
__global__ __launch_bounds__(256) void k_s1(const float* __restrict__ pred,
    const __bf16* __restrict__ WaP, const __bf16* __restrict__ embP,
    float* __restrict__ WaS, float* __restrict__ embOut) {
  __shared__ __align__(16) __bf16 As[16 * OO];   // 16KB: padded 16x512 A matrix
  const int tid = threadIdx.x, lane = tid & 31, w = tid >> 5;
  const int sel = blockIdx.x >> 2;
  float vals[16];
#pragma unroll
  for (int j = 0; j < 16; ++j) vals[j] = pred[w * OO + lane + 32 * j];
  if (sel == 0) {
#pragma unroll
    for (int j = 0; j < 16; ++j)
      As[w * OO + lane + 32 * j] = (__bf16)(1.0f / (1.0f + __expf(-vals[j])));
  } else {
    float m = vals[0];
#pragma unroll
    for (int j = 1; j < 16; ++j) m = fmaxf(m, vals[j]);
    m = wave_max(m);
    float s = 0.0f, ev[16];
#pragma unroll
    for (int j = 0; j < 16; ++j) { ev[j] = __expf(vals[j] - m); s += ev[j]; }
    s = wave_sum(s);
    const float inv = 1.0f / s;
#pragma unroll
    for (int j = 0; j < 16; ++j)
      As[w * OO + lane + 32 * j] = (__bf16)(ev[j] * inv);
  }
#pragma unroll
  for (int j = 0; j < 16; ++j) As[(w + 8) * OO + lane + 32 * j] = (__bf16)0.0f;
  __syncthreads();

  const __bf16* Bp = sel ? embP : WaP;
  float* Dst = sel ? embOut : WaS;
  const int nt = (blockIdx.x & 3) * 8 + w;
  const int n = (nt << 4) + (lane & 15);
  const int r = lane & 15;
  v8f acc = {};
#pragma unroll 4
  for (int k0 = 0; k0 < OO; k0 += 32) {
    v16bf a = load_a_frag(&As[r * OO], k0, lane);
    v16bf b = load_b_packed(Bp, k0, nt, lane);
    acc = wmma_bf16(a, b, acc);
  }
#pragma unroll
  for (int i = 0; i < 8; ++i) {
    const int m = ((lane >> 4) << 3) + i;
    Dst[m * OO + n] = acc[i];
  }
}

// ---------------- S2: scores[b,t'] = sum_o tanh(UaH + WaS)*Va ----------------
// one wave per (b,t') row, 1024 rows; 8 rows per block share batch index b, so
// WaS[b,:] and Va are staged once per block into LDS via async copies.
__global__ __launch_bounds__(256) void k_s2(const float* __restrict__ UaH,
    const float* __restrict__ WaS, const float* __restrict__ Va,
    float* __restrict__ scores) {
  __shared__ __align__(16) float sWaS[OO];  // 2KB
  __shared__ __align__(16) float sVa[OO];   // 2KB
  const int tid = threadIdx.x, lane = tid & 31, w = tid >> 5;
  const int row = blockIdx.x * 8 + w;       // 0..1023
  const int b = blockIdx.x >> 4;            // == row >> 7 (8 rows/block)
  if (tid < 128) {
    async_copy_b128(lds_off_of(&sWaS[tid * 4]), WaS + b * OO + tid * 4);
  } else {
    const int q = tid - 128;
    async_copy_b128(lds_off_of(&sVa[q * 4]), Va + q * 4);
  }
  wait_async0();
  __syncthreads();

  const float* rowp = UaH + (size_t)row * OO;
  __builtin_prefetch(rowp, 0, 3);           // global_prefetch_b8 on the hot row
  float sum = 0.0f;
#pragma unroll 4
  for (int j = 0; j < 16; ++j) {
    const int o = lane + 32 * j;
    const float u = rowp[o] + sWaS[o];
    // tanh(u) = 1 - 2/(exp(2u)+1): one v_exp_f32 + one reciprocal
    const float e2 = __expf(2.0f * u);
    sum += (1.0f - 2.0f / (e2 + 1.0f)) * sVa[o];
  }
  sum = wave_sum(sum);
  if (lane == 0) scores[row] = sum;
}

// ---------------- S3: WoY; softmax over T; ctx = sm @ inputs (to bf16, padded) ----
__global__ __launch_bounds__(256) void k_s3(const float* __restrict__ scores,
    const float* __restrict__ inputs, const float* __restrict__ embOut,
    const float* __restrict__ Wo, float* __restrict__ WoY,
    __bf16* __restrict__ ctx_bf) {
  __shared__ float red[256];
  __shared__ float sm[128];
  const int b = blockIdx.x, tid = threadIdx.x;
  // WoY[b] = sum_o embOut[b,o] * Wo[o]
  red[tid] = embOut[b * OO + tid] * Wo[tid] +
             embOut[b * OO + 256 + tid] * Wo[256 + tid];
  __syncthreads();
  for (int s = 128; s > 0; s >>= 1) { if (tid < s) red[tid] += red[tid + s]; __syncthreads(); }
  if (tid == 0) WoY[b] = red[0];
  __syncthreads();
  // softmax over t' = 0..127
  const float sc = (tid < 128) ? scores[b * 128 + tid] : -3.0e38f;
  red[tid] = sc; __syncthreads();
  for (int s = 128; s > 0; s >>= 1) { if (tid < s) red[tid] = fmaxf(red[tid], red[tid + s]); __syncthreads(); }
  const float mx = red[0]; __syncthreads();
  const float e = (tid < 128) ? __expf(sc - mx) : 0.0f;
  red[tid] = e; __syncthreads();
  for (int s = 128; s > 0; s >>= 1) { if (tid < s) red[tid] += red[tid + s]; __syncthreads(); }
  const float tot = red[0]; __syncthreads();
  if (tid < 128) sm[tid] = e / tot;
  __syncthreads();
  // ctx[b,d] = sum_t sm[t] * inputs[b,t,d]  (coalesced over d)
#pragma unroll
  for (int dd = 0; dd < 2; ++dd) {
    const int d = tid + dd * 256;
    float acc = 0.0f;
#pragma unroll 4
    for (int t = 0; t < 128; ++t)
      acc += sm[t] * inputs[((size_t)(b * TT + t)) * DD + d];
    ctx_bf[b * DD + d] = (__bf16)acc;
    ctx_bf[(b + 8) * DD + d] = (__bf16)0.0f;   // zero M-padding rows
  }
}

// ---------------- S4: pred = out_base(t) + ctx@Co + WoY ; write out & pred ------
// 32 wave-tiles (M=16 padded x N=512). ctx tile (16KB) async-staged to LDS once
// per block; every wave's A-fragments then come from LDS.
__global__ __launch_bounds__(256) void k_s4(const __bf16* __restrict__ ctx_bf,
    const __bf16* __restrict__ CoP, const float* __restrict__ WoY,
    float* __restrict__ out, float* __restrict__ pred, int t) {
  __shared__ __align__(16) __bf16 Actx[16 * DD];   // 16KB
  const int tid = threadIdx.x, lane = tid & 31, w = tid >> 5;
#pragma unroll
  for (int c = 0; c < 4; ++c) {
    const int chunk = tid + c * 256;               // 1024 x 16B chunks
    async_copy_b128(lds_off_of(&Actx[chunk * 8]), ctx_bf + chunk * 8);
  }
  wait_async0();
  __syncthreads();

  const int nt = blockIdx.x * 8 + w;        // 0..31
  const int n = (nt << 4) + (lane & 15);
  const int r = lane & 15;
  v8f acc = {};
#pragma unroll 4
  for (int k0 = 0; k0 < DD; k0 += 32) {
    v16bf a = load_a_frag(&Actx[r * DD], k0, lane);
    v16bf b = load_b_packed(CoP, k0, nt, lane);
    acc = wmma_bf16(a, b, acc);
  }
  if (lane < 16) {                          // rows 0..7 are real batch rows
#pragma unroll
    for (int i = 0; i < 8; ++i) {
      const size_t idx = ((size_t)(i * TT + t)) * OO + n;
      const float v = acc[i] + out[idx] + WoY[i];
      out[idx] = v;                         // final output for step t
      pred[i * OO + n] = v;                 // feeds step t+1
    }
  }
}

// ---------------- workspace layout (bytes; all 256-aligned) ----------------
constexpr size_t OFF_INP  = 0;                          // B*T*D bf16  (1,048,576)
constexpr size_t OFF_UA   = OFF_INP + 1048576;          // packed D*O bf16 (524,288)
constexpr size_t OFF_UO   = OFF_UA + 524288;
constexpr size_t OFF_WA   = OFF_UO + 524288;
constexpr size_t OFF_EMB  = OFF_WA + 524288;
constexpr size_t OFF_CO   = OFF_EMB + 524288;
constexpr size_t OFF_UAH  = OFF_CO + 524288;            // B*T*O f32   (2,097,152)
constexpr size_t OFF_WAS  = OFF_UAH + 2097152;          // 16*O f32
constexpr size_t OFF_EOUT = OFF_WAS + 32768;            // 16*O f32
constexpr size_t OFF_CTX  = OFF_EOUT + 32768;           // 16*D bf16
constexpr size_t OFF_SC   = OFF_CTX + 16384;            // B*T f32
constexpr size_t OFF_WOY  = OFF_SC + 4096;              // 8 f32 (padded)
constexpr size_t OFF_PRED = OFF_WOY + 256;              // B*O f32
// total ~5.6 MB

extern "C" void kernel_launch(void* const* d_in, const int* in_sizes, int n_in,
                              void* d_out, int out_size, void* d_ws, size_t ws_size,
                              hipStream_t stream) {
  const float* inputs = (const float*)d_in[0];
  const float* Wa  = (const float*)d_in[1];
  const float* Ua  = (const float*)d_in[2];
  const float* Va  = (const float*)d_in[3];
  const float* Ba  = (const float*)d_in[4];
  const float* Wo  = (const float*)d_in[5];
  const float* Uo  = (const float*)d_in[6];
  const float* Co  = (const float*)d_in[7];
  const float* Bo  = (const float*)d_in[8];
  const float* emb = (const float*)d_in[9];
  float* out = (float*)d_out;
  char* ws = (char*)d_ws;

  __bf16* inp_bf = (__bf16*)(ws + OFF_INP);
  __bf16* UaP    = (__bf16*)(ws + OFF_UA);
  __bf16* UoP    = (__bf16*)(ws + OFF_UO);
  __bf16* WaP    = (__bf16*)(ws + OFF_WA);
  __bf16* embP   = (__bf16*)(ws + OFF_EMB);
  __bf16* CoP    = (__bf16*)(ws + OFF_CO);
  float*  UaH    = (float*)(ws + OFF_UAH);
  float*  WaS    = (float*)(ws + OFF_WAS);
  float*  embOut = (float*)(ws + OFF_EOUT);
  __bf16* ctx_bf = (__bf16*)(ws + OFF_CTX);
  float*  scores = (float*)(ws + OFF_SC);
  float*  WoY    = (float*)(ws + OFF_WOY);
  float*  pred   = (float*)(ws + OFF_PRED);

  // one-time (per call) conversions / fragment packing + init
  k_cvt<<<256, 256, 0, stream>>>(inputs, inp_bf, BB * TT * DD);
  k_pack<<<64, 256, 0, stream>>>(Ua,  UaP,  DD);
  k_pack<<<64, 256, 0, stream>>>(Uo,  UoP,  DD);
  k_pack<<<64, 256, 0, stream>>>(Wa,  WaP,  OO);
  k_pack<<<64, 256, 0, stream>>>(emb, embP, OO);
  k_pack<<<64, 256, 0, stream>>>(Co,  CoP,  DD);
  k_zero<<<1, 256, 0, stream>>>(pred, BB * OO);

  // big WMMA precompute: UaH = inputs@Ua + Ba ; out = roll(inputs)@Uo + Bo
  k_pre<<<512, 256, 0, stream>>>(inp_bf, UaP, UoP, Ba, Bo, UaH, out);

  // sequential scan: stream order carries the step-to-step dependency
  for (int t = 0; t < TT; ++t) {
    k_s1<<<8,   256, 0, stream>>>(pred, WaP, embP, WaS, embOut);
    k_s2<<<128, 256, 0, stream>>>(UaH, WaS, Va, scores);
    k_s3<<<8,   256, 0, stream>>>(scores, inputs, embOut, Wo, WoY, ctx_bf);
    k_s4<<<4,   256, 0, stream>>>(ctx_bf, CoP, WoY, out, pred, t);
  }
}
